// StateSpace2dJoints_54417235640681
// MI455X (gfx1250) — compile-verified, matrix operand order checked by reading, not run on hardware
//
#include <hip/hip_runtime.h>
#include <hip/hip_bf16.h>

typedef __bf16 bf16_t;
typedef __attribute__((ext_vector_type(16))) __bf16 v16bf;
typedef __attribute__((ext_vector_type(8)))  float  v8f;

#define B_TOT 8192
#define T_TOT 75
#define NSEED 10
#define NPRED 65
#define D_IN  34
#define E_DIM 128
#define H_DIM 64
#define G_DIM 256   // 4*H
#define KX    64    // encoder K padded 34 -> 64
#define DPAD  48    // decoder N padded 34 -> 48

// Fragment load: each lane reads 16 contiguous bf16 along K.
// lane 0-15 -> row (lane), K = k0..k0+15 ; lane 16-31 -> same rows, K = k0+16..k0+31
__device__ __forceinline__ v16bf frag_ld(const bf16_t* base, int stride, int lane, int k0) {
    const bf16_t* p = base + (lane & 15) * stride + (k0 + ((lane >> 4) << 4));
    return *(const v16bf*)p;
}

__device__ __forceinline__ v8f wmma_bf16(v16bf a, v16bf b, v8f c) {
    return __builtin_amdgcn_wmma_f32_16x16x32_bf16(false, a, false, b, (short)0, c,
                                                   false, false);
}

__device__ __forceinline__ float sigmoidf_(float x) {
    return 1.0f / (1.0f + __expf(-x));
}

// One LSTM cell step for a 16-row batch tile. gates[b][n] layout: i|f|g|o blocks of 64.
// Hidden block hb covers cols 16*hb..16*hb+15 of each gate -> tiles {hb, 4+hb, 8+hb, 12+hb}.
template <bool WARM>
__device__ __forceinline__ void lstm_step(const bf16_t* __restrict__ embT,
                                          bf16_t* __restrict__ hT,
                                          const bf16_t* __restrict__ sWih,
                                          const bf16_t* __restrict__ sWhh,
                                          v8f cfr[4], const float gbias[16],
                                          int lane, int col, int rhalf) {
    v16bf ha0 = frag_ld(hT, H_DIM, lane, 0);
    v16bf ha1 = frag_ld(hT, H_DIM, lane, 32);
    v16bf ea[4];
    if (WARM) {
#pragma unroll
        for (int kk = 0; kk < 4; ++kk) ea[kk] = frag_ld(embT, E_DIM, lane, 32 * kk);
    }
#pragma unroll
    for (int hb = 0; hb < 4; ++hb) {
        v8f g4[4];
#pragma unroll
        for (int gi = 0; gi < 4; ++gi) {
            const int nt = gi * 4 + hb;             // gate tile index (0..15)
            v8f acc = {};
            const bf16_t* whhB = &sWhh[nt * 16 * H_DIM];
            acc = wmma_bf16(ha0, frag_ld(whhB, H_DIM, lane, 0), acc);
            acc = wmma_bf16(ha1, frag_ld(whhB, H_DIM, lane, 32), acc);
            if (WARM) {
                const bf16_t* wihB = &sWih[nt * 16 * E_DIM];
#pragma unroll
                for (int kk = 0; kk < 4; ++kk)
                    acc = wmma_bf16(ea[kk], frag_ld(wihB, E_DIM, lane, 32 * kk), acc);
            }
#pragma unroll
            for (int r = 0; r < 8; ++r) acc[r] += gbias[nt];
            g4[gi] = acc;
        }
        v8f cn = cfr[hb];
        v8f hn = {};
#pragma unroll
        for (int r = 0; r < 8; ++r) {
            float ig = sigmoidf_(g4[0][r]);
            float fg = sigmoidf_(g4[1][r]);
            float gg = tanhf(g4[2][r]);
            float og = sigmoidf_(g4[3][r]);
            float cc = fg * cn[r] + ig * gg;
            cn[r] = cc;
            hn[r] = og * tanhf(cc);
        }
        cfr[hb] = cn;
#pragma unroll
        for (int r = 0; r < 8; ++r)
            hT[(rhalf + r) * H_DIM + hb * 16 + col] = (bf16_t)hn[r];
    }
    __syncthreads();
}

__global__ __launch_bounds__(128, 1)
void lstm_rollout_kernel(const float* __restrict__ x,
                         const float* __restrict__ W_enc, const float* __restrict__ b_enc,
                         const float* __restrict__ W_ih,  const float* __restrict__ W_hh,
                         const float* __restrict__ b_ih,  const float* __restrict__ b_hh,
                         const float* __restrict__ W_dec, const float* __restrict__ b_dec,
                         float* __restrict__ out) {
    // Weights as B-operands: [N][K] bf16 (gates[b][n] = sum_k in[b][k] * W[n][k]).
    __shared__ __align__(32) bf16_t sWih[G_DIM * E_DIM];   // 64 KB
    __shared__ __align__(32) bf16_t sWhh[G_DIM * H_DIM];   // 32 KB
    __shared__ __align__(32) bf16_t sWenc[E_DIM * KX];     // 16 KB (K zero-padded)
    __shared__ __align__(32) bf16_t sWdec[DPAD * H_DIM];   //  6 KB (N zero-padded)
    __shared__ __align__(32) bf16_t sX[4][16 * KX];        //  8 KB per-wave x tile
    __shared__ __align__(32) bf16_t sEmb[4][16 * E_DIM];   // 16 KB per-wave emb tile
    __shared__ __align__(32) bf16_t sH[4][16 * H_DIM];     //  8 KB per-wave h tile

    const int tid   = threadIdx.x;
    const int lane  = tid & 31;
    const int wave  = tid >> 5;
    const int b0    = blockIdx.x * 64 + wave * 16;
    const int col   = lane & 15;           // N column within a 16-wide tile
    const int rhalf = (lane >> 4) * 8;     // D-fragment row offset

    // ---- convert weights f32 -> bf16 into LDS (cooperative) ----
    for (int i = tid; i < G_DIM * E_DIM; i += 128) sWih[i] = (bf16_t)W_ih[i];
    for (int i = tid; i < G_DIM * H_DIM; i += 128) sWhh[i] = (bf16_t)W_hh[i];
    for (int i = tid; i < E_DIM * KX; i += 128) {
        int e = i / KX, k = i % KX;
        sWenc[i] = (k < D_IN) ? (bf16_t)W_enc[e * D_IN + k] : (bf16_t)0.0f;
    }
    for (int i = tid; i < DPAD * H_DIM; i += 128) {
        int d = i / H_DIM, k = i % H_DIM;
        sWdec[i] = (d < D_IN) ? (bf16_t)W_dec[d * H_DIM + k] : (bf16_t)0.0f;
    }
    for (int i = lane; i < 16 * KX;    i += 32) sX[wave][i] = (bf16_t)0.0f; // pad region
    for (int i = lane; i < 16 * H_DIM; i += 32) sH[wave][i] = (bf16_t)0.0f; // h0 = 0
    __syncthreads();

    // ---- per-lane hoisted biases ----
    float gbias[16];
#pragma unroll
    for (int nt = 0; nt < 16; ++nt) {
        int n = nt * 16 + col;
        gbias[nt] = b_ih[n] + b_hh[n];
    }
    float ebias[8];
#pragma unroll
    for (int e = 0; e < 8; ++e) ebias[e] = b_enc[e * 16 + col];
    float dbias[3];
#pragma unroll
    for (int dt = 0; dt < 3; ++dt) {
        int d = dt * 16 + col;
        dbias[dt] = (d < D_IN) ? b_dec[d] : 0.0f;
    }

    // ---- persistent fragment state ----
    v8f cfr[4];
#pragma unroll
    for (int t = 0; t < 4; ++t) cfr[t] = (v8f){};
    v8f ofr[3];                                   // cumulative output, init x[:, 9, :]
#pragma unroll
    for (int dt = 0; dt < 3; ++dt) {
        v8f o = {};
#pragma unroll
        for (int r = 0; r < 8; ++r) {
            int c = dt * 16 + col;
            o[r] = (c < D_IN)
                 ? x[((long)(b0 + rhalf + r) * T_TOT + (NSEED - 1)) * D_IN + c] : 0.0f;
        }
        ofr[dt] = o;
    }

    bf16_t* xT   = sX[wave];
    bf16_t* embT = sEmb[wave];
    bf16_t* hT   = sH[wave];

    // ================= warm-up: 10 encoded steps =================
    for (int t = 0; t < NSEED; ++t) {
        {   // load + convert x[:, t, 0:34] into padded bf16 tile (2 lanes per row)
            int r = lane >> 1, half = lane & 1;
            const float* xr = &x[((long)(b0 + r) * T_TOT + t) * D_IN];
            for (int c = half * 17; c < half * 17 + 17; ++c)
                xT[r * KX + c] = (bf16_t)xr[c];
        }
        __syncthreads();

        // emb = relu(x @ W_enc^T + b_enc), written to LDS as next A-operand
        v16bf xa0 = frag_ld(xT, KX, lane, 0);
        v16bf xa1 = frag_ld(xT, KX, lane, 32);
#pragma unroll
        for (int e = 0; e < 8; ++e) {
            v8f acc = {};
            const bf16_t* wB = &sWenc[(e * 16) * KX];
            acc = wmma_bf16(xa0, frag_ld(wB, KX, lane, 0), acc);
            acc = wmma_bf16(xa1, frag_ld(wB, KX, lane, 32), acc);
#pragma unroll
            for (int r = 0; r < 8; ++r) {
                float v = acc[r] + ebias[e];
                v = v > 0.0f ? v : 0.0f;
                embT[(rhalf + r) * E_DIM + e * 16 + col] = (bf16_t)v;
            }
        }
        __syncthreads();

        lstm_step<true>(embT, hT, sWih, sWhh, cfr, gbias, lane, col, rhalf);
    }

    // ================= rollout: 65 autoregressive steps =================
    for (int s = 0; s < NPRED; ++s) {
        lstm_step<false>(embT, hT, sWih, sWhh, cfr, gbias, lane, col, rhalf);

        // dec = h @ W_dec^T + b_dec ; out_s = dec + out_{s-1}
        v16bf ha0 = frag_ld(hT, H_DIM, lane, 0);
        v16bf ha1 = frag_ld(hT, H_DIM, lane, 32);
#pragma unroll
        for (int dt = 0; dt < 3; ++dt) {
            v8f acc = {};
            const bf16_t* wB = &sWdec[(dt * 16) * H_DIM];
            acc = wmma_bf16(ha0, frag_ld(wB, H_DIM, lane, 0), acc);
            acc = wmma_bf16(ha1, frag_ld(wB, H_DIM, lane, 32), acc);
            v8f o = ofr[dt];
            int c = dt * 16 + col;
#pragma unroll
            for (int r = 0; r < 8; ++r) {
                o[r] += acc[r] + dbias[dt];
                if (c < D_IN)
                    out[((long)(b0 + rhalf + r) * NPRED + s) * D_IN + c] = o[r];
            }
            ofr[dt] = o;
        }
    }
}

extern "C" void kernel_launch(void* const* d_in, const int* in_sizes, int n_in,
                              void* d_out, int out_size, void* d_ws, size_t ws_size,
                              hipStream_t stream) {
    const float* x     = (const float*)d_in[0];
    const float* W_enc = (const float*)d_in[1];
    const float* b_enc = (const float*)d_in[2];
    const float* W_ih  = (const float*)d_in[3];
    const float* W_hh  = (const float*)d_in[4];
    const float* b_ih  = (const float*)d_in[5];
    const float* b_hh  = (const float*)d_in[6];
    const float* W_dec = (const float*)d_in[7];
    const float* b_dec = (const float*)d_in[8];
    float* out = (float*)d_out;

    dim3 grid(B_TOT / 64);   // 128 workgroups, 64 batch rows each (16 per wave)
    dim3 block(128);         // 4 waves of 32
    lstm_rollout_kernel<<<grid, block, 0, stream>>>(x, W_enc, b_enc, W_ih, W_hh,
                                                    b_ih, b_hh, W_dec, b_dec, out);
}
